// G_Synthesis_ResNet_21251498181274
// MI455X (gfx1250) — matrix operand order repaired and checked by reading, not compiled, not run on hardware
//
#include <hip/hip_runtime.h>
#include <math.h>

typedef __attribute__((ext_vector_type(16))) _Float16 v16h;
typedef __attribute__((ext_vector_type(8)))  _Float16 v8h;
typedef __attribute__((ext_vector_type(8)))  float    v8f;

#define SQRT2F 1.41421356237f
#define INV_SQRT2F 0.70710678118f

// ---------------------------------------------------------------------------
// affine: s[b,c] = (style[b,:] . aff_w[c,:]) / sqrt(SDIM) + aff_b[c]
// ---------------------------------------------------------------------------
__global__ void affine_kernel(const float* __restrict__ style,
                              const float* __restrict__ aw,
                              const float* __restrict__ ab,
                              float* __restrict__ s,
                              _Float16* __restrict__ s16,
                              int C, int SDIM) {
  int idx = blockIdx.x * blockDim.x + threadIdx.x;
  if (idx >= 16 * C) return;
  int b = idx / C, c = idx % C;
  const float* st = style + (size_t)b * SDIM;
  const float* w  = aw + (size_t)c * SDIM;
  float acc = 0.f;
  for (int k = 0; k < SDIM; ++k) acc += st[k] * w[k];
  float v = acc * rsqrtf((float)SDIM) + ab[c];
  s[idx] = v;
  s16[idx] = (_Float16)v;
}

// ---------------------------------------------------------------------------
// demod[b,cout] = rsqrt(scale2 * sum_{cin,taps} (W * s[b,cin])^2 + 1e-8)
// ---------------------------------------------------------------------------
__global__ void demod_kernel(const float* __restrict__ W,
                             const float* __restrict__ s,
                             float* __restrict__ dm,
                             int Cin, int Cout, int so, int si, float scale2) {
  int idx = blockIdx.x * blockDim.x + threadIdx.x;
  if (idx >= 16 * Cout) return;
  int b = idx / Cout, co = idx % Cout;
  float acc = 0.f;
  for (int ci = 0; ci < Cin; ++ci) {
    float sv = s[b * Cin + ci];
    const float* wp = W + (size_t)co * so + (size_t)ci * si;
    float ws = 0.f;
    #pragma unroll
    for (int t = 0; t < 9; ++t) { float w = wp[t]; ws += w * w; }
    acc += ws * sv * sv;
  }
  dm[idx] = rsqrtf(scale2 * acc + 1e-8f);
}

// ---------------------------------------------------------------------------
// Pack weights into per-lane A-fragment order (contiguous k, same order as the B loads):
// frag = (mt*nchunks + cc)*ntaps + t ; P[frag*512 + lane*16 + h]
// m = lane&15, k = ((lane>>4)<<4) + h, c = cc*32 + k
// ---------------------------------------------------------------------------
__global__ void pack_w_kernel(const float* __restrict__ W,
                              _Float16* __restrict__ P,
                              int Cout, int Cin, int ntaps,
                              int ws_o, int ws_i, int ws_t, float wscale) {
  int idx = blockIdx.x * blockDim.x + threadIdx.x;
  long total = (long)Cout * Cin * ntaps;
  if (idx >= total) return;
  const int nchunks = Cin >> 5;
  int e = idx & 511; int fb = idx >> 9;
  int lane = e >> 4, h = e & 15;
  int t  = fb % ntaps;
  int cc = (fb / ntaps) % nchunks;
  int mt = fb / (ntaps * nchunks);
  int m = (mt << 4) + (lane & 15);
  int k = ((lane >> 4) << 4) + h;
  int c = (cc << 5) + k;
  P[idx] = (_Float16)(wscale * W[(size_t)m * ws_o + (size_t)c * ws_i + (size_t)t * ws_t]);
}

// ---------------------------------------------------------------------------
// Fused upsample kernels: E6 = blur4 (*) W3, split into 4 parity 3x3 kernels,
// written directly in packed-fragment f16 order. Wt layout [Cin][Cout][3][3].
// ---------------------------------------------------------------------------
__global__ void build_ep_kernel(const float* __restrict__ Wt,
                                _Float16* __restrict__ P,
                                int Cin, int Cout, float scale) {
  int idx = blockIdx.x * blockDim.x + threadIdx.x;
  if (idx >= Cin * Cout) return;
  int ci = idx / Cout, co = idx % Cout;
  const float* w = Wt + ((size_t)ci * Cout + co) * 9;
  const float bk1[4] = {0.25f, 0.75f, 0.75f, 0.25f};
  float e6[6][6];
  #pragma unroll
  for (int r = 0; r < 6; ++r)
    #pragma unroll
    for (int c = 0; c < 6; ++c) {
      float v = 0.f;
      #pragma unroll
      for (int j = 0; j < 4; ++j) {
        int ky = r - 3 + j;
        if (ky < 0 || ky > 2) continue;
        #pragma unroll
        for (int i = 0; i < 4; ++i) {
          int kx = c - 3 + i;
          if (kx < 0 || kx > 2) continue;
          v += bk1[j] * bk1[i] * w[ky * 3 + kx];
        }
      }
      e6[r][c] = v;
    }
  const int nchunks = Cin >> 5;
  const int mt = co >> 4;
  const int cc = ci >> 5;
  const int k  = ci & 31;
  const int lane = (co & 15) | ((k >> 4) << 4);
  const int h    = k & 15;
  const size_t par_stride = (size_t)Cout * Cin * 9;
  for (int py = 0; py < 2; ++py)
    for (int px = 0; px < 2; ++px) {
      size_t base = (size_t)(py * 2 + px) * par_stride;
      #pragma unroll
      for (int ky = 0; ky < 3; ++ky)
        #pragma unroll
        for (int kx = 0; kx < 3; ++kx) {
          int t = ky * 3 + kx;
          size_t frag = ((size_t)mt * nchunks + cc) * 9 + t;
          P[base + frag * 512 + lane * 16 + h] =
              (_Float16)(scale * e6[4 - 2 * ky + py][4 - 2 * kx + px]);
        }
    }
}

// ---------------------------------------------------------------------------
// Implicit-GEMM conv via V_WMMA_F32_16X16X32_F16, NHWC f16 activations,
// packed fragment-order f16 weights.  One wave computes 16(cout) x
// 16*NTILES(pixel) with A-fragment reuse across the NTILES accumulators.
// NTAPS/NTILES compile-time: tap loop fully unrolled, constant (dy,dx).
// flags: 1 = demod[b,cout], 2 = bias + leaky_relu(0.2)*sqrt(2)
// ---------------------------------------------------------------------------
template <int NTAPS, int NTILES>
__global__ __launch_bounds__(32)
void conv_wmma_kernel(const _Float16* __restrict__ in,
                      const _Float16* __restrict__ Wp,
                      const _Float16* __restrict__ s16,
                      const float* __restrict__ dm,
                      const float* __restrict__ bias,
                      _Float16* __restrict__ out,
                      int Cin, int Cout, int lw, int nchunks,
                      int Hob, int Wob, int y0, int x0, int ystep, int xstep,
                      int flags) {
  const int lane = threadIdx.x;            // 0..31, wave32
  const int b    = blockIdx.z;
  const int m0   = blockIdx.y << 4;        // cout tile base
  const int p0   = blockIdx.x * (16 * NTILES);
  const int H    = 1 << lw;                // square spatial
  const int kb   = (lane >> 4) << 4;       // B-fragment K base for this lane

  int ty[NTILES], tx[NTILES];
  #pragma unroll
  for (int u = 0; u < NTILES; ++u) {
    int p = p0 + u * 16 + (lane & 15);
    ty[u] = p >> lw;
    tx[u] = p & (H - 1);
  }

  const v16h* wpw = (const v16h*)Wp + ((size_t)blockIdx.y * nchunks * NTAPS) * 32 + lane;
  const _Float16* in_b = in + (((size_t)b << (2 * lw)) * Cin);
  const _Float16* sp = s16 ? (s16 + b * Cin + kb) : nullptr;

  v8f acc[NTILES];
  v8f zero = {};
  #pragma unroll
  for (int u = 0; u < NTILES; ++u) acc[u] = zero;

  for (int cc = 0; cc < nchunks; ++cc) {
    const int c0 = cc << 5;
    v16h sv;
    if (sp) sv = *(const v16h*)(sp + c0);
    __builtin_prefetch(wpw + 32 * NTAPS, 0, 1);   // next chunk's A stream
    #pragma unroll
    for (int t = 0; t < NTAPS; ++t) {
      const int dy = (NTAPS == 9) ? (t / 3 - 1) : 0;
      const int dx = (NTAPS == 9) ? (t % 3 - 1) : 0;
      v16h afrag = wpw[t * 32];                   // contiguous 32B per lane
      #pragma unroll
      for (int u = 0; u < NTILES; ++u) {
        const int iy = ty[u] + dy, ix = tx[u] + dx;
        const bool inb = (iy >= 0) & (iy < H) & (ix >= 0) & (ix < H);
        v16h bfrag = {};
        if (inb)
          bfrag = *(const v16h*)(in_b + (((size_t)iy << lw) + ix) * Cin + c0 + kb);
        if (sp) bfrag = bfrag * sv;               // v_pk_mul_f16, co-executes w/ WMMA
        acc[u] = __builtin_amdgcn_wmma_f32_16x16x32_f16(
            false, afrag, false, bfrag, (short)0, acc[u], false, false);
      }
    }
    wpw += 32 * NTAPS;
  }

  // ---- epilogue: demod, bias, lrelu, one v8h NHWC store per lane per tile
  const int mb = m0 + ((lane >> 4) << 3);
  #pragma unroll
  for (int u = 0; u < NTILES; ++u) {
    const int oy = y0 + ty[u] * ystep;
    const int ox = x0 + tx[u] * xstep;
    v8h vst;
    #pragma unroll
    for (int r = 0; r < 8; ++r) {
      const int co = mb + r;
      float v = acc[u][r];
      if (flags & 1) v *= dm[b * Cout + co];
      if (flags & 2) {
        v += bias[co];
        v = (v > 0.f ? v : 0.2f * v) * SQRT2F;
      }
      vst[r] = (_Float16)v;
    }
    *(v8h*)(out + (((size_t)b * Hob + oy) * Wob + ox) * Cout + mb) = vst;
  }
}

// ---------------------------------------------------------------------------
// h_next = (nf2 + blurup2(z)) / sqrt(2).  NHWC.
// ---------------------------------------------------------------------------
__global__ void skip_blur_add_kernel(const _Float16* __restrict__ nf2,
                                     const _Float16* __restrict__ z,
                                     _Float16* __restrict__ hout,
                                     int C, int H2, int W2) {
  size_t idx = (size_t)blockIdx.x * blockDim.x + threadIdx.x;
  size_t total = (size_t)16 * C * H2 * W2;
  if (idx >= total) return;
  int c  = idx % C;  size_t r0 = idx / C;
  int xw = r0 % W2;  size_t r1 = r0 / W2;
  int yh = r1 % H2;  int b = (int)(r1 / H2);
  const int H = H2 >> 1, Wd = W2 >> 1;
  const float bk1[4] = {0.25f, 0.75f, 0.75f, 0.25f};
  float acc = 0.f;
  #pragma unroll
  for (int j = 0; j < 4; ++j) {
    int uy = yh - 2 + j;
    if (uy & 1) continue;
    int ih = uy >> 1;
    if (ih < 0 || ih >= H) continue;
    #pragma unroll
    for (int i = 0; i < 4; ++i) {
      int ux = xw - 2 + i;
      if (ux & 1) continue;
      int iw = ux >> 1;
      if (iw < 0 || iw >= Wd) continue;
      acc += bk1[j] * bk1[i] * (float)z[(((size_t)b * H + ih) * Wd + iw) * C + c];
    }
  }
  hout[idx] = (_Float16)(((float)nf2[idx] + acc) * INV_SQRT2F);
}

// ---------------------------------------------------------------------------
// final 1x1 modconv 32->3 (no demod) + bias; NHWC f16 in, NCHW f32 out
// ---------------------------------------------------------------------------
__global__ void out_layer_kernel(const _Float16* __restrict__ h,
                                 const float* __restrict__ W,
                                 const float* __restrict__ s,
                                 const float* __restrict__ bias,
                                 float* __restrict__ out, int Cin, int HW) {
  size_t idx = (size_t)blockIdx.x * blockDim.x + threadIdx.x;
  size_t total = (size_t)16 * 3 * HW;
  if (idx >= total) return;
  int pp = idx % HW; size_t r0 = idx / HW;
  int co = r0 % 3;   int b = (int)(r0 / 3);
  const float sc = rsqrtf((float)Cin);
  const _Float16* hp = h + ((size_t)b * HW + pp) * Cin;
  float acc = 0.f;
  for (int ci = 0; ci < Cin; ++ci)
    acc += (float)hp[ci] * s[b * Cin + ci] * W[co * Cin + ci];
  out[idx] = acc * sc + bias[co];
}

// init_feat NCHW f32 [16][512][4][4] -> NHWC f16
__global__ void cvt_init_kernel(const float* __restrict__ in,
                                _Float16* __restrict__ out) {
  int idx = blockIdx.x * blockDim.x + threadIdx.x;
  if (idx >= 16 * 512 * 16) return;
  int x = idx & 3, y = (idx >> 2) & 3, c = (idx >> 4) & 511, b = idx >> 13;
  out[((((size_t)b << 2) + y) * 4 + x) * 512 + c] = (_Float16)in[idx];
}

// ---------------------------------------------------------------------------
extern "C" void kernel_launch(void* const* d_in, const int* in_sizes, int n_in,
                              void* d_out, int out_size, void* d_ws, size_t ws_size,
                              hipStream_t stream) {
  (void)in_sizes; (void)n_in; (void)out_size; (void)ws_size;
  const float* init_feat = (const float*)d_in[0];
  const float* style     = (const float*)d_in[1];
  const int Bn = 16, SDIM = 512;
  const int chans[7] = {512, 512, 512, 256, 128, 64, 32};

  const size_t ACT = (size_t)16 * 256 * 256 * 32;        // 33.55M halfs each
  _Float16* act0 = (_Float16*)d_ws;
  _Float16* act1 = act0 + ACT;
  _Float16* act2 = act1 + ACT;
  _Float16* acts[3] = {act0, act1, act2};
  _Float16* zbuf  = act2 + ACT;
  _Float16* wpack = zbuf + (size_t)16 * 128 * 128 * 32;
  float* s0f = (float*)(wpack + (size_t)4 * 512 * 512 * 9);
  float* s1f = s0f + Bn * 512;
  float* dm  = s1f + Bn * 512;
  _Float16* s0h = (_Float16*)(dm + Bn * 512);
  _Float16* s1h = s0h + Bn * 512;

  auto cdiv = [](long a, long b) { return (int)((a + b - 1) / b); };

  auto launch_conv = [&](const _Float16* inp, const _Float16* wp, const _Float16* sh,
                         const float* dmp, const float* bp, _Float16* outp,
                         int Cin, int Cout, int lwv, int ntaps,
                         int Hob, int Wob, int y0, int x0, int ystep, int xstep,
                         int flags) {
    const int npix = 1 << (2 * lwv);
    if (npix >= 64) {
      dim3 g(npix / 64, Cout / 16, Bn);
      if (ntaps == 9)
        conv_wmma_kernel<9, 4><<<g, 32, 0, stream>>>(inp, wp, sh, dmp, bp, outp,
            Cin, Cout, lwv, Cin / 32, Hob, Wob, y0, x0, ystep, xstep, flags);
      else
        conv_wmma_kernel<1, 4><<<g, 32, 0, stream>>>(inp, wp, sh, dmp, bp, outp,
            Cin, Cout, lwv, Cin / 32, Hob, Wob, y0, x0, ystep, xstep, flags);
    } else {
      dim3 g(npix / 16, Cout / 16, Bn);
      if (ntaps == 9)
        conv_wmma_kernel<9, 1><<<g, 32, 0, stream>>>(inp, wp, sh, dmp, bp, outp,
            Cin, Cout, lwv, Cin / 32, Hob, Wob, y0, x0, ystep, xstep, flags);
      else
        conv_wmma_kernel<1, 1><<<g, 32, 0, stream>>>(inp, wp, sh, dmp, bp, outp,
            Cin, Cout, lwv, Cin / 32, Hob, Wob, y0, x0, ystep, xstep, flags);
    }
  };

  // ---------------- first layer: modconv3x3 512->512 @4x4, demod, lrelu
  {
    const float* aw = (const float*)d_in[2];
    const float* ab = (const float*)d_in[3];
    const float* w  = (const float*)d_in[4];
    const float* actb = (const float*)d_in[5];
    affine_kernel<<<cdiv(Bn * 512, 256), 256, 0, stream>>>(style, aw, ab, s0f, s0h, 512, SDIM);
    demod_kernel<<<cdiv(Bn * 512, 256), 256, 0, stream>>>(w, s0f, dm, 512, 512,
                                                          512 * 9, 9, 1.f / (512.f * 9.f));
    pack_w_kernel<<<cdiv((long)512 * 512 * 9, 256), 256, 0, stream>>>(
        w, wpack, 512, 512, 9, 512 * 9, 9, 1, 1.f / sqrtf(512.f * 9.f));
    cvt_init_kernel<<<cdiv(Bn * 512 * 16, 256), 256, 0, stream>>>(init_feat, act0);
    launch_conv(act0, wpack, s0h, dm, actb, act1, 512, 512, 2, 9, 4, 4, 0, 0, 1, 1, 3);
  }

  int hidx = 1, H = 4, lw = 2;
  for (int j = 0; j < 6; ++j) {
    const int cin = chans[j], cout = chans[j + 1];
    const int base = 6 + 9 * j;
    const float* up_aff_w  = (const float*)d_in[base + 0];
    const float* up_aff_b  = (const float*)d_in[base + 1];
    const float* up_w      = (const float*)d_in[base + 2];
    const float* up_act_b  = (const float*)d_in[base + 3];
    const float* btk_aff_w = (const float*)d_in[base + 4];
    const float* btk_aff_b = (const float*)d_in[base + 5];
    const float* btk_w     = (const float*)d_in[base + 6];
    const float* btk_act_b = (const float*)d_in[base + 7];
    const float* skip_w    = (const float*)d_in[base + 8];
    const int t1 = (hidx + 1) % 3, t2 = (hidx + 2) % 3;
    const int H2 = 2 * H;
    const size_t par_stride = (size_t)cout * cin * 9;

    // ---- fused upsample modconv (tconv k3 s2 + FIR blur) as 4 parity convs
    affine_kernel<<<cdiv(Bn * cin, 256), 256, 0, stream>>>(
        style + (size_t)(2 * j + 1) * Bn * SDIM, up_aff_w, up_aff_b, s0f, s0h, cin, SDIM);
    demod_kernel<<<cdiv(Bn * cout, 256), 256, 0, stream>>>(
        up_w, s0f, dm, cin, cout, 9, cout * 9, 1.f / (cin * 9.f));
    build_ep_kernel<<<cdiv((long)cin * cout, 256), 256, 0, stream>>>(
        up_w, wpack, cin, cout, 1.f / sqrtf(cin * 9.f));
    for (int pp = 0; pp < 4; ++pp)
      launch_conv(acts[hidx], wpack + (size_t)pp * par_stride, s0h, dm, up_act_b,
                  acts[t1], cin, cout, lw, 9, H2, H2, pp >> 1, pp & 1, 2, 2, 3);

    // ---- bottleneck modconv3x3 cout->cout @2H, demod, lrelu
    affine_kernel<<<cdiv(Bn * cout, 256), 256, 0, stream>>>(
        style + (size_t)(2 * j + 2) * Bn * SDIM, btk_aff_w, btk_aff_b, s1f, s1h, cout, SDIM);
    demod_kernel<<<cdiv(Bn * cout, 256), 256, 0, stream>>>(
        btk_w, s1f, dm, cout, cout, cout * 9, 9, 1.f / (cout * 9.f));
    pack_w_kernel<<<cdiv((long)cout * cout * 9, 256), 256, 0, stream>>>(
        btk_w, wpack, cout, cout, 9, cout * 9, 9, 1, 1.f / sqrtf(cout * 9.f));
    launch_conv(acts[t1], wpack, s1h, dm, btk_act_b, acts[t2],
                cout, cout, lw + 1, 9, H2, H2, 0, 0, 1, 1, 3);

    // ---- skip: plain 1x1 eq-conv (WMMA GEMM) then blur-upsample + add
    pack_w_kernel<<<cdiv((long)cout * cin, 256), 256, 0, stream>>>(
        skip_w, wpack, cout, cin, 1, 1, cout, 1, 1.f / sqrtf((float)cin));
    launch_conv(acts[hidx], wpack, nullptr, nullptr, nullptr, zbuf,
                cin, cout, lw, 1, H, H, 0, 0, 1, 1, 0);
    skip_blur_add_kernel<<<cdiv((long)Bn * cout * H2 * H2, 256), 256, 0, stream>>>(
        acts[t2], zbuf, acts[t1], cout, H2, H2);

    hidx = t1;
    H = H2;
    lw += 1;
  }

  // ---------------- out layer: modconv1x1 32->3 (no demod) + bias
  {
    const float* aw = (const float*)d_in[60];
    const float* ab = (const float*)d_in[61];
    const float* w  = (const float*)d_in[62];
    const float* bias = (const float*)d_in[63];
    affine_kernel<<<cdiv(Bn * 32, 256), 256, 0, stream>>>(
        style + (size_t)13 * Bn * SDIM, aw, ab, s1f, s1h, 32, SDIM);
    out_layer_kernel<<<cdiv((long)Bn * 3 * H * H, 256), 256, 0, stream>>>(
        acts[hidx], w, s1f, bias, (float*)d_out, 32, H * H);
  }
}
